// HistAtt2_27848567947646
// MI455X (gfx1250) — matrix operand order repaired and checked by reading, not compiled, not run on hardware
//
#include <hip/hip_runtime.h>
#include <hip/hip_bf16.h>
#include <cstdint>
#include <cstddef>

// ---------- native (trivial) vector types: safe inside unions ----------
typedef __attribute__((ext_vector_type(16))) _Float16 v16h;
typedef __attribute__((ext_vector_type(8)))  float    v8f;
typedef __attribute__((ext_vector_type(4)))  float    f32x4;
typedef __attribute__((ext_vector_type(4)))  unsigned u32x4;
typedef __attribute__((ext_vector_type(2)))  unsigned u32x2;

#define BATCH   2048
#define LH      200      // history length
#define DD      64       // embedding dim
#define IN_DIM  256      // 4*D feature dim
#define NT1     13       // GEMM1 N tiles (200 -> 208)
#define KT1     8        // GEMM1 K steps (256 / 32)
#define NT2     5        // GEMM2 N tiles (80)
#define KT2     7        // GEMM2 K steps (224 / 32), K padded 200 -> 224
#define H1STR   224      // H1 row stride (halves)
#define W1TILES (KT1*NT1)        // 104
#define W2TILES (KT2*NT2)        // 35
#define W1HALVES (W1TILES*512)   // 53248 halves

// =====================================================================
// Kernel 0: convert W1/W2 (f32, row-major [K][N]) into f16 tiles laid out
// as this file's WMMA B operand (contiguous k order, matching A):
//   lane j of a wave owns column n = (j&15), K-half (j>>4):
//   16 contiguous halves = K = kbase + 0..15, stored contiguously per lane.
// One 32-thread block per 32x16 tile.
// =====================================================================
__global__ __launch_bounds__(32) void prep_weights(
    const float* __restrict__ W1, const float* __restrict__ W2,
    _Float16* __restrict__ ws)
{
    const int tile = blockIdx.x;
    const int lane = threadIdx.x;
    const int nloc = lane & 15;
    const int khalf = lane >> 4;
    if (tile < W1TILES) {
        const int kt = tile / NT1, nt = tile % NT1;
        _Float16* dst = ws + tile * 512 + lane * 16;
        const int n = nt * 16 + nloc;            // 0..207 (pad >=200)
        const int kb = kt * 32 + khalf * 16;     // 0..255
        #pragma unroll
        for (int i = 0; i < 16; ++i) {
            float v = (n < 200) ? W1[(kb + i) * 200 + n] : 0.0f;
            dst[i] = (_Float16)v;
        }
    } else {
        const int t2 = tile - W1TILES;
        const int kt = t2 / NT2, nt = t2 % NT2;
        _Float16* dst = ws + W1HALVES + t2 * 512 + lane * 16;
        const int n = nt * 16 + nloc;            // 0..79
        const int kb = kt * 32 + khalf * 16;     // 0..223 (pad >=200)
        #pragma unroll
        for (int i = 0; i < 16; ++i) {
            const int k = kb + i;
            float v = (k < 200) ? W2[k * 80 + n] : 0.0f;
            dst[i] = (_Float16)v;
        }
    }
}

// =====================================================================
// Kernel 1: per-wave MLP scorer. One wave = one (batch b, 16-row L chunk).
// 4 waves per block, 8KB LDS scratch per wave (features -> H1 -> H2).
// =====================================================================
__global__ __launch_bounds__(128) void hist_att_scores(
    const float* __restrict__ x_item,
    const float* __restrict__ user_hist,
    const int*   __restrict__ hist_len,
    const _Float16* __restrict__ ws,
    const float* __restrict__ b1v,
    const float* __restrict__ b2v,
    const float* __restrict__ W3,
    const float* __restrict__ b3v,
    float* __restrict__ score_out)
{
    __shared__ __align__(16) _Float16 smem[4][4096];   // 8KB per wave
    const int lane = threadIdx.x & 31;
    const int warp = threadIdx.x >> 5;
    const int task = blockIdx.x * 4 + warp;            // 0 .. 2048*13-1
    const int b     = task / NT1;
    const int chunk = task - b * NT1;
    const int l0 = chunk * 16;
    _Float16* F = &smem[warp][0];

    // ---------------- Phase 1: build feature tile [16][256] f16 ----------
    const float* xb  = x_item + b * DD;
    const float* hbp = user_hist + (size_t)b * (LH * DD);
    #pragma unroll
    for (int r = 0; r < 8; ++r) {
        const int idx = r * 128 + lane * 4;   // covers 16 rows * 64 cols
        const int row = idx >> 6;
        const int col = idx & 63;
        const int l = l0 + row;
        f32x4 h4 = (f32x4)0.0f;
        if (l < LH) h4 = *(const f32x4*)(hbp + l * DD + col);
        const f32x4 x4 = *(const f32x4*)(xb + col);
        const f32x4 m4 = x4 * h4;
        const f32x4 s4 = x4 - h4;
        union { _Float16 h[4]; u32x2 u; } t0, t1, t2, t3;
        #pragma unroll
        for (int i = 0; i < 4; ++i) {
            t0.h[i] = (_Float16)x4[i];
            t1.h[i] = (_Float16)h4[i];
            t2.h[i] = (_Float16)m4[i];
            t3.h[i] = (_Float16)s4[i];
        }
        _Float16* fr = F + row * IN_DIM;
        *(u32x2*)(fr + col)       = t0.u;   // x tile
        *(u32x2*)(fr + 64 + col)  = t1.u;   // hist
        *(u32x2*)(fr + 128 + col) = t2.u;   // x*hist
        *(u32x2*)(fr + 192 + col) = t3.u;   // x-hist
    }
    asm volatile("s_wait_dscnt 0" ::: "memory");
    __builtin_amdgcn_wave_barrier();

    // ---------------- Phase 2: GEMM1  [16x256] x [256x208] -------------
    const int m  = lane & 15;    // A row / C col-in-tile
    const int hh = lane >> 4;    // K-half selector
    v8f acc[NT1];
    #pragma unroll
    for (int i = 0; i < NT1; ++i) acc[i] = (v8f)0.0f;
    const u32x4* wq = (const u32x4*)ws;     // 64 u32x4 per tile
    for (int kt = 0; kt < KT1; ++kt) {
        union { v16h v; u32x4 q[2]; } A;
        const _Float16* ap = F + m * IN_DIM + kt * 32 + 16 * hh;
        A.q[0] = *(const u32x4*)ap;           // K = 16h + 0..7
        A.q[1] = *(const u32x4*)(ap + 8);     // K = 16h + 8..15
        #pragma unroll
        for (int nt = 0; nt < NT1; ++nt) {
            union { v16h v; u32x4 q[2]; } Bt;
            const u32x4* bp = wq + (kt * NT1 + nt) * 64 + lane * 2;
            Bt.q[0] = bp[0];
            Bt.q[1] = bp[1];
            acc[nt] = __builtin_amdgcn_wmma_f32_16x16x32_f16(
                false, A.v, false, Bt.v, (short)0, acc[nt], false, false);
        }
    }
    asm volatile("s_wait_dscnt 0" ::: "memory");
    __builtin_amdgcn_wave_barrier();

    // ---------------- Phase 3: bias+ReLU -> H1 f16 [16][224] (reuse F) --
    _Float16* H1 = F;
    for (int i = lane; i < 16 * 24; i += 32) {     // zero K-pad cols 200..223
        const int rr = i / 24;
        const int cc = 200 + (i - rr * 24);
        H1[rr * H1STR + cc] = (_Float16)0.0f;
    }
    #pragma unroll
    for (int nt = 0; nt < NT1; ++nt) {
        const int n = nt * 16 + m;
        const float bias = (n < 200) ? b1v[n] : 0.0f;
        #pragma unroll
        for (int r = 0; r < 8; ++r) {
            const float v = fmaxf(acc[nt][r] + bias, 0.0f);
            if (n < 200) H1[(r + 8 * hh) * H1STR + n] = (_Float16)v;
        }
    }
    asm volatile("s_wait_dscnt 0" ::: "memory");
    __builtin_amdgcn_wave_barrier();

    // ---------------- Phase 4: GEMM2  [16x224] x [224x80] ---------------
    v8f acc2[NT2];
    #pragma unroll
    for (int i = 0; i < NT2; ++i) acc2[i] = (v8f)0.0f;
    const u32x4* w2q = wq + (W1HALVES / 8);
    for (int kt = 0; kt < KT2; ++kt) {
        union { v16h v; u32x4 q[2]; } A;
        const _Float16* ap = H1 + m * H1STR + kt * 32 + 16 * hh;
        A.q[0] = *(const u32x4*)ap;
        A.q[1] = *(const u32x4*)(ap + 8);
        #pragma unroll
        for (int nt = 0; nt < NT2; ++nt) {
            union { v16h v; u32x4 q[2]; } Bt;
            const u32x4* bp = w2q + (kt * NT2 + nt) * 64 + lane * 2;
            Bt.q[0] = bp[0];
            Bt.q[1] = bp[1];
            acc2[nt] = __builtin_amdgcn_wmma_f32_16x16x32_f16(
                false, A.v, false, Bt.v, (short)0, acc2[nt], false, false);
        }
    }
    asm volatile("s_wait_dscnt 0" ::: "memory");
    __builtin_amdgcn_wave_barrier();

    // ---------------- Phase 5: bias+ReLU -> H2 f32 [16][80] (reuse F) ---
    float* H2 = (float*)F;
    #pragma unroll
    for (int nt = 0; nt < NT2; ++nt) {
        const int n = nt * 16 + m;
        const float bias = b2v[n];
        #pragma unroll
        for (int r = 0; r < 8; ++r)
            H2[(r + 8 * hh) * 80 + n] = fmaxf(acc2[nt][r] + bias, 0.0f);
    }
    asm volatile("s_wait_dscnt 0" ::: "memory");
    __builtin_amdgcn_wave_barrier();

    // ---------------- Phase 6: final dot + mask -> scores ---------------
    if (lane < 16) {
        const int l = l0 + lane;
        if (l < LH) {
            float s = b3v[0];
            for (int c = 0; c < 80; ++c) s += H2[lane * 80 + c] * W3[c];
            const float sc = (l < hist_len[b]) ? s : -1024.0f;  // NULL_ATTENTION
            score_out[(size_t)b * LH + l] = sc;
        }
    }
}

// =====================================================================
// Kernel 2: per-b softmax over L (masked entries carry -1024, NOT -inf,
// so hist_len==0 correctly yields uniform 1/200) + attention-weighted sums.
// =====================================================================
__global__ __launch_bounds__(256) void hist_att_softmax(
    const float* __restrict__ user_hist,
    const float* __restrict__ user_click,
    const float* __restrict__ score,
    float* __restrict__ rep,
    float* __restrict__ crep)
{
    const int b = blockIdx.x;
    const int t = threadIdx.x;
    __shared__ float sc[256];
    __shared__ float red[16];
    __shared__ float part[4][DD][2];

    const float v = (t < LH) ? score[(size_t)b * LH + t] : -3.0e38f;

    // block max (8 waves of 32)
    float mx = v;
    #pragma unroll
    for (int off = 16; off > 0; off >>= 1) mx = fmaxf(mx, __shfl_xor(mx, off, 32));
    if ((t & 31) == 0) red[t >> 5] = mx;
    __syncthreads();
    float M = red[0];
    #pragma unroll
    for (int i = 1; i < 8; ++i) M = fmaxf(M, red[i]);

    const float e = (t < LH) ? __expf(v - M) : 0.0f;
    sc[t] = e;
    float sm = e;
    #pragma unroll
    for (int off = 16; off > 0; off >>= 1) sm += __shfl_xor(sm, off, 32);
    if ((t & 31) == 0) red[8 + (t >> 5)] = sm;
    __syncthreads();
    float S = 0.0f;
    #pragma unroll
    for (int i = 0; i < 8; ++i) S += red[8 + i];
    const float inv = 1.0f / S;

    // weighted sums: 4 partial groups over l, 64 lanes per group over d
    const int d = t & 63;
    const int p = t >> 6;
    const float* hb = user_hist  + (size_t)b * (LH * DD) + d;
    const float* cb = user_click + (size_t)b * (LH * DD) + d;
    float s1 = 0.0f, s2 = 0.0f;
    for (int l = p; l < LH; l += 4) {
        const float a = sc[l] * inv;
        s1 += a * hb[l * DD];
        s2 += a * cb[l * DD];
    }
    part[p][d][0] = s1;
    part[p][d][1] = s2;
    __syncthreads();
    if (t < DD) {
        const float r1 = part[0][t][0] + part[1][t][0] + part[2][t][0] + part[3][t][0];
        const float r2 = part[0][t][1] + part[1][t][1] + part[2][t][1] + part[3][t][1];
        rep [(size_t)b * DD + t] = r1;
        crep[(size_t)b * DD + t] = r2;
    }
}

// =====================================================================
extern "C" void kernel_launch(void* const* d_in, const int* in_sizes, int n_in,
                              void* d_out, int out_size, void* d_ws, size_t ws_size,
                              hipStream_t stream)
{
    const float* x_item    = (const float*)d_in[0];
    const float* user_hist = (const float*)d_in[1];
    const int*   hist_len  = (const int*)  d_in[2];
    const float* user_clk  = (const float*)d_in[3];
    const float* W1        = (const float*)d_in[4];
    const float* b1        = (const float*)d_in[5];
    const float* W2        = (const float*)d_in[6];
    const float* b2        = (const float*)d_in[7];
    const float* W3        = (const float*)d_in[8];
    const float* b3        = (const float*)d_in[9];

    float* out   = (float*)d_out;
    float* rep   = out;                       // [2048,64]
    float* crep  = out + BATCH * DD;          // [2048,64]
    float* score = out + 2 * BATCH * DD;      // [2048,200]
    _Float16* ws = (_Float16*)d_ws;           // 142336 B of prepped weights

    // 1) pre-swizzle weights to f16 WMMA-B layout (deterministic each call)
    prep_weights<<<W1TILES + W2TILES, 32, 0, stream>>>(W1, W2, ws);

    // 2) WMMA MLP scorer: 2048*13 wave tasks, 4 waves/block
    const int nblocks = (BATCH * NT1) / 4;    // 6656
    hist_att_scores<<<nblocks, 128, 0, stream>>>(
        x_item, user_hist, hist_len, ws, b1, b2, W3, b3, score);

    // 3) softmax + attention-weighted reductions: one block per batch row
    hist_att_softmax<<<BATCH, 256, 0, stream>>>(
        user_hist, user_clk, score, rep, crep);
}
